// MixtureOfExperts_13383118094605
// MI455X (gfx1250) — matrix operand order repaired and checked by reading, not compiled, not run on hardware
//
#include <hip/hip_runtime.h>
#include <math.h>

// ---------------------------------------------------------------------------
// MoE (DeepSeek-style) for MI455X / gfx1250, wave32, WMMA bf16 path.
//
// Roofline: ~450MB fp32 weight/activation traffic vs 116 GFLOP => HBM-bound
// (~19us floor @ 23.3 TB/s). Strategy: single-pass weight streaming (b128
// loads feeding v_wmma_f32_16x16x32_bf16), index-list dispatch (no [E,C,H]
// scatter buffers), and async global->LDS DMA for gathered activation tiles.
// ---------------------------------------------------------------------------

typedef __attribute__((ext_vector_type(16))) __bf16 v16bf;
typedef __attribute__((ext_vector_type(8)))  float  v8f;
typedef __attribute__((__vector_size__(4 * sizeof(int)))) int v4i;

#define T_TOK 4096
#define HDIM  1024
#define IDIM  512
#define NEXP  64
#define TOPK  8
#define CAP   1024
#define MT    32     // token rows per block tile
#define KC    128    // K chunk of A staged into LDS (fp32)

// ---- async global->LDS (gfx1250 ASYNCcnt path) ----------------------------
#if defined(__has_builtin)
#  if __has_builtin(__builtin_amdgcn_global_load_async_to_lds_b128)
#    define ASYNC_MODE 1
#  else
#    define ASYNC_MODE 0
#  endif
#  if __has_builtin(__builtin_amdgcn_s_wait_asynccnt)
#    define WAIT_ASYNC() __builtin_amdgcn_s_wait_asynccnt(0)
#  else
#    define WAIT_ASYNC() asm volatile("s_wait_asynccnt 0x0" ::: "memory")
#  endif
#else
#  define ASYNC_MODE 0
#  define WAIT_ASYNC() asm volatile("s_wait_asynccnt 0x0" ::: "memory")
#endif

__device__ __forceinline__ void async_copy16(const float* gsrc, float* ldst) {
#if ASYNC_MODE == 1
  // global_load_async_to_lds_b128: 16B HBM->LDS DMA, no VGPR round-trip.
  __builtin_amdgcn_global_load_async_to_lds_b128((v4i*)gsrc, (v4i*)ldst, 0, 0);
#else
  float4 f = *reinterpret_cast<const float4*>(gsrc);
  reinterpret_cast<float4*>(ldst)[0] = f;
#endif
}

__device__ __forceinline__ void async_fence() {
#if ASYNC_MODE
  WAIT_ASYNC();
#endif
}

__device__ __forceinline__ v8f zero8() {
  v8f z;
#pragma unroll
  for (int q = 0; q < 8; ++q) z[q] = 0.0f;
  return z;
}

__device__ __forceinline__ v8f wmma_bf16(v16bf a, v16bf b, v8f c) {
  // emits v_wmma_f32_16x16x32_bf16
  return __builtin_amdgcn_wmma_f32_16x16x32_bf16(false, a, false, b,
                                                 (short)0, c, false, false);
}

// A-fragment (16x32 bf16, lane = row M=lane%16):
//   lanes 0-15 : elems 0..7 -> K=0..7,   elems 8..15 -> K=16..23
//   lanes 16-31: elems 0..7 -> K=8..15,  elems 8..15 -> K=24..31
// fp32 source in LDS, convert to bf16 at fragment load (amortized over tiles).
__device__ __forceinline__ v16bf load_a_lds_f32(const float* row, int kbase, int half) {
  v16bf a;
  const float* p0 = row + kbase + 8 * half;
  const float* p1 = row + kbase + 16 + 8 * half;
#pragma unroll
  for (int i = 0; i < 8; ++i) { a[i] = (__bf16)p0[i]; a[i + 8] = (__bf16)p1[i]; }
  return a;
}

// bf16 source in LDS (phase-2 intermediate h).
__device__ __forceinline__ v16bf load_a_lds_bf16(const __bf16* row, int kbase, int half) {
  v16bf a;
  const __bf16* p0 = row + kbase + 8 * half;
  const __bf16* p1 = row + kbase + 16 + 8 * half;
#pragma unroll
  for (int i = 0; i < 8; ++i) { a[i] = p0[i]; a[i + 8] = p1[i]; }
  return a;
}

// B-fragment (32x16 bf16, lane L holds column N=L%16; elems 0..7 -> K=8*(L/16)+i,
// elems 8..15 -> K=16+8*(L/16)+i), fp32 weight row (row-major in K).
__device__ __forceinline__ v16bf load_b_gl(const float* col, int kbase, int half) {
  const float4* p0 = reinterpret_cast<const float4*>(col + kbase + 8 * half);
  const float4* p1 = reinterpret_cast<const float4*>(col + kbase + 16 + 8 * half);
  v16bf b;
#pragma unroll
  for (int q = 0; q < 4; ++q) {
    float4 f = (q < 2) ? p0[q] : p1[q - 2];
    b[4 * q + 0] = (__bf16)f.x; b[4 * q + 1] = (__bf16)f.y;
    b[4 * q + 2] = (__bf16)f.z; b[4 * q + 3] = (__bf16)f.w;
  }
  return b;
}

// ---------------------------------------------------------------------------
// Kernel 1: sigmoid gating, top-8 + normalize, atomic slot dispatch.
// ---------------------------------------------------------------------------
__global__ __launch_bounds__(256)
void gate_topk_kernel(const float* __restrict__ x, const float* __restrict__ gw,
                      int* __restrict__ counts, int* __restrict__ tok_list,
                      float* __restrict__ w_list) {
  __shared__ float xs[HDIM];
  __shared__ float sc[NEXP];
  const int t = blockIdx.x;
  const int tid = threadIdx.x;
  const int lane = tid & 31, wid = tid >> 5;

  for (int k = tid; k < HDIM; k += 256) xs[k] = x[(size_t)t * HDIM + k];
  __syncthreads();

#pragma unroll
  for (int j = 0; j < 8; ++j) {
    const int e = wid * 8 + j;
    const float* w = gw + (size_t)e * HDIM;
    float p = 0.0f;
    for (int k = lane; k < HDIM; k += 32) p = fmaf(xs[k], w[k], p);
#pragma unroll
    for (int off = 16; off > 0; off >>= 1) p += __shfl_xor(p, off, 32);
    if (lane == 0) sc[e] = 1.0f / (1.0f + __expf(-p));
  }
  __syncthreads();

  if (tid == 0) {
    int idx[TOPK]; float wv[TOPK]; float sum = 0.0f;
#pragma unroll
    for (int k = 0; k < TOPK; ++k) {
      float best = -1.0f; int bi = 0;
      for (int e = 0; e < NEXP; ++e) { float s = sc[e]; if (s > best) { best = s; bi = e; } }
      idx[k] = bi; wv[k] = best; sum += best; sc[bi] = -2.0f;
    }
    const float inv = 1.0f / sum;
#pragma unroll
    for (int k = 0; k < TOPK; ++k) {
      const int e = idx[k];
      const int slot = atomicAdd(&counts[e], 1);
      if (slot < CAP) {
        tok_list[e * CAP + slot] = t;
        w_list[e * CAP + slot]   = wv[k] * inv;
      }
    }
  }
}

// ---------------------------------------------------------------------------
// Kernel 2: fused SwiGLU grouped-GEMM. Blocks [0, E*CAP/MT) = routed experts
// (gathered rows, weighted atomic combine); remaining T/MT blocks = shared
// expert (identity rows, weight 1). A tiles staged via async global->LDS DMA.
// ---------------------------------------------------------------------------
__global__ __launch_bounds__(256)
void moe_swiglu_kernel(const float* __restrict__ x,
                       const float* __restrict__ ewg, const float* __restrict__ ewu,
                       const float* __restrict__ ewd,
                       const float* __restrict__ swg, const float* __restrict__ swu,
                       const float* __restrict__ swd,
                       const int* __restrict__ tok_list, const float* __restrict__ w_list,
                       const int* __restrict__ counts, float* __restrict__ out) {
  __shared__ float  a_sh[MT][KC];     // gathered activation chunk (fp32, DMA'd)
  __shared__ __bf16 h_sh[MT][IDIM];   // silu(g)*u intermediate, bf16
  __shared__ int    tok_sh[MT];
  __shared__ float  w_sh[MT];

  const int bid  = blockIdx.x;
  const int tid  = threadIdx.x;
  const int lane = tid & 31;
  const int wid  = tid >> 5;
  const int half = lane >> 4;
  const int ln   = lane & 15;

  const bool shared_e = bid >= NEXP * (CAP / MT);
  const float *wg, *wu, *wd;
  int m0, cnt, eid = 0;
  if (shared_e) {
    m0 = (bid - NEXP * (CAP / MT)) * MT; cnt = T_TOK;
    wg = swg; wu = swu; wd = swd;
  } else {
    eid = bid / (CAP / MT);
    m0  = (bid % (CAP / MT)) * MT;
    cnt = counts[eid]; if (cnt > CAP) cnt = CAP;
    wg = ewg + (size_t)eid * IDIM * HDIM;
    wu = ewu + (size_t)eid * IDIM * HDIM;
    wd = ewd + (size_t)eid * HDIM * IDIM;
  }
  if (m0 >= cnt) return;   // uniform per block

  if (tid < MT) {
    const int slot = m0 + tid;
    int tok = 0; float w = 0.0f;
    if (slot < cnt) {
      if (shared_e) { tok = slot; w = 1.0f; }
      else { tok = tok_list[eid * CAP + slot]; w = w_list[eid * CAP + slot]; }
    }
    tok_sh[tid] = tok; w_sh[tid] = w;
  }
  __syncthreads();

  const int rg  = wid & 1;    // row group (0/1) of the 32-row tile
  const int cg0 = wid >> 1;   // column-group phase (0..3)
  const int am  = rg * 16 + ln;

  // staging geometry: 32 rows x KC floats, 8 threads/row, 4 x b128 per thread
  const int st_r  = tid >> 3;
  const int st_c0 = (tid & 7) * (KC / 8);

  // ---- phase 1: g = A*Wg^T, u = A*Wu^T (A gathered via index list) ----
  v8f accg[8], accu[8];
#pragma unroll
  for (int i = 0; i < 8; ++i) { accg[i] = zero8(); accu[i] = zero8(); }

  for (int kc = 0; kc < HDIM / KC; ++kc) {
    { // gather chunk into LDS: async DMA (ASYNCcnt) when available
      const float* src = x + (size_t)tok_sh[st_r] * HDIM + kc * KC + st_c0;
#pragma unroll
      for (int q = 0; q < KC / 8 / 4; ++q)
        async_copy16(src + 4 * q, &a_sh[st_r][st_c0 + 4 * q]);
      async_fence();
    }
    __syncthreads();

    for (int ks = 0; ks < KC / 32; ++ks) {
      const v16bf afrag = load_a_lds_f32(&a_sh[am][0], ks * 32, half);
      const int kg = kc * KC + ks * 32;
#pragma unroll
      for (int it = 0; it < 8; ++it) {
        const int n = (cg0 + 4 * it) * 16 + ln;
        const float* gcol = wg + (size_t)n * HDIM;
        const float* ucol = wu + (size_t)n * HDIM;
        __builtin_prefetch(gcol + kg + 64, 0, 1);  // global_prefetch_b8
        const v16bf bg = load_b_gl(gcol, kg, half);
        const v16bf bu = load_b_gl(ucol, kg, half);
        accg[it] = wmma_bf16(afrag, bg, accg[it]);
        accu[it] = wmma_bf16(afrag, bu, accu[it]);
      }
    }
    __syncthreads();
  }

  // silu(g) * u -> h_sh (C/D layout: VGPR v -> M = v + 8*half, N = ln)
#pragma unroll
  for (int it = 0; it < 8; ++it) {
    const int cg = cg0 + 4 * it;
#pragma unroll
    for (int v = 0; v < 8; ++v) {
      const int row = rg * 16 + v + 8 * half;
      const int col = cg * 16 + ln;
      const float g = accg[it][v], u = accu[it][v];
      const float s = g / (1.0f + __expf(-g));
      h_sh[row][col] = (__bf16)(s * u);
    }
  }
  __syncthreads();

  // ---- phase 2: out += w_row * (h * Wd^T), 64 column groups over H ----
  for (int it = 0; it < 16; ++it) {
    const int cg = cg0 + 4 * it;           // 0..63
    const int n  = cg * 16 + ln;           // output column in H
    const float* dcol = wd + (size_t)n * IDIM;
    v8f acc = zero8();
    for (int ks = 0; ks < IDIM / 32; ++ks) {
      const v16bf afrag = load_a_lds_bf16(&h_sh[am][0], ks * 32, half);
      const v16bf bfrag = load_b_gl(dcol, ks * 32, half);
      acc = wmma_bf16(afrag, bfrag, acc);
    }
#pragma unroll
    for (int v = 0; v < 8; ++v) {
      const int row = rg * 16 + v + 8 * half;
      const float w = w_sh[row];
      if (w != 0.0f)
        atomicAdd(&out[(size_t)tok_sh[row] * HDIM + n], acc[v] * w);
    }
  }
}

// ---------------------------------------------------------------------------
extern "C" void kernel_launch(void* const* d_in, const int* in_sizes, int n_in,
                              void* d_out, int out_size, void* d_ws, size_t ws_size,
                              hipStream_t stream) {
  const float* x      = (const float*)d_in[0];
  const float* gate_w = (const float*)d_in[1];
  const float* swg    = (const float*)d_in[2];
  const float* swu    = (const float*)d_in[3];
  const float* swd    = (const float*)d_in[4];
  const float* ewg    = (const float*)d_in[5];
  const float* ewu    = (const float*)d_in[6];
  const float* ewd    = (const float*)d_in[7];
  float* out = (float*)d_out;

  // workspace layout: [counts: E ints][tok_list: E*CAP ints][w_list: E*CAP floats]
  int*   counts   = (int*)d_ws;
  int*   tok_list = counts + NEXP;
  float* w_list   = (float*)(tok_list + NEXP * CAP);

  (void)hipMemsetAsync(counts, 0, NEXP * sizeof(int), stream);
  (void)hipMemsetAsync(out, 0, (size_t)T_TOK * HDIM * sizeof(float), stream);

  gate_topk_kernel<<<T_TOK, 256, 0, stream>>>(x, gate_w, counts, tok_list, w_list);

  const int routed_blocks = NEXP * (CAP / MT);   // 2048 (most early-exit)
  const int shared_blocks = T_TOK / MT;          // 128
  moe_swiglu_kernel<<<routed_blocks + shared_blocks, 256, 0, stream>>>(
      x, ewg, ewu, ewd, swg, swu, swd, tok_list, w_list, counts, out);
}
